// BitLinear_43525198578345
// MI455X (gfx1250) — 8-bit matrix operand order repaired and checked by reading, compiled, not run on hardware
//
#include <hip/hip_runtime.h>
#include <stdint.h>

typedef int v8i __attribute__((ext_vector_type(8)));
typedef int v4i __attribute__((ext_vector_type(4)));
typedef int v2i __attribute__((ext_vector_type(2)));

#define D_IN   2048
#define D_OUT  8192
#define M_TOT  16384

__global__ void wabs_reduce_kernel(const float4* __restrict__ w4,
                                   float* __restrict__ wsum, int n4) {
    __shared__ float red[256];
    float s = 0.0f;
    for (int i = blockIdx.x * blockDim.x + threadIdx.x; i < n4;
         i += gridDim.x * blockDim.x) {
        float4 v = w4[i];
        s += fabsf(v.x) + fabsf(v.y) + fabsf(v.z) + fabsf(v.w);
    }
    red[threadIdx.x] = s;
    __syncthreads();
    for (int o = 128; o > 0; o >>= 1) {
        if (threadIdx.x < o) red[threadIdx.x] += red[threadIdx.x + o];
        __syncthreads();
    }
    if (threadIdx.x == 0) atomicAdd(wsum, red[0]);
}

__global__ void ternarize_kernel(const float4* __restrict__ w4,
                                 const float* __restrict__ wsum,
                                 char* __restrict__ wq, int n4) {
    const float scale = fmaxf(*wsum * (1.0f / (float)((long)D_OUT * D_IN)), 1e-8f);
    const float inv = 1.0f / scale;
    int i = blockIdx.x * blockDim.x + threadIdx.x;
    if (i >= n4) return;
    float4 v = w4[i];
    int q0 = (int)fminf(fmaxf(rintf(v.x * inv), -1.0f), 1.0f);
    int q1 = (int)fminf(fmaxf(rintf(v.y * inv), -1.0f), 1.0f);
    int q2 = (int)fminf(fmaxf(rintf(v.z * inv), -1.0f), 1.0f);
    int q3 = (int)fminf(fmaxf(rintf(v.w * inv), -1.0f), 1.0f);
    unsigned p = (q0 & 255u) | ((q1 & 255u) << 8) | ((q2 & 255u) << 16)
               | ((unsigned)(q3 & 255) << 24);
    *(int*)(wq + (size_t)i * 4) = (int)p;
}

__global__ void rmsq_kernel(const float* __restrict__ x,
                            const float* __restrict__ gamma,
                            char* __restrict__ xq,
                            float* __restrict__ xscale) {
    __shared__ float red[256];
    const int row = blockIdx.x;
    const int t   = threadIdx.x;
    const float4* xr = (const float4*)(x + (size_t)row * D_IN);
    const float4* g4 = (const float4*)gamma;

    float4 v0 = xr[2 * t], v1 = xr[2 * t + 1];
    float ss = v0.x*v0.x + v0.y*v0.y + v0.z*v0.z + v0.w*v0.w
             + v1.x*v1.x + v1.y*v1.y + v1.z*v1.z + v1.w*v1.w;
    red[t] = ss;
    __syncthreads();
    for (int o = 128; o > 0; o >>= 1) {
        if (t < o) red[t] += red[t + o];
        __syncthreads();
    }
    const float inv = rsqrtf(red[0] * (1.0f / (float)D_IN) + 1e-6f);
    __syncthreads();

    float4 g0 = g4[2 * t], g1 = g4[2 * t + 1];
    float xn[8];
    xn[0] = v0.x * inv * g0.x; xn[1] = v0.y * inv * g0.y;
    xn[2] = v0.z * inv * g0.z; xn[3] = v0.w * inv * g0.w;
    xn[4] = v1.x * inv * g1.x; xn[5] = v1.y * inv * g1.y;
    xn[6] = v1.z * inv * g1.z; xn[7] = v1.w * inv * g1.w;

    float am = 0.0f;
    #pragma unroll
    for (int i = 0; i < 8; i++) am = fmaxf(am, fabsf(xn[i]));
    red[t] = am;
    __syncthreads();
    for (int o = 128; o > 0; o >>= 1) {
        if (t < o) red[t] = fmaxf(red[t], red[t + o]);
        __syncthreads();
    }
    const float amax = fmaxf(red[0], 1e-5f);
    const float qs = 127.0f / amax;

    int q[8];
    #pragma unroll
    for (int i = 0; i < 8; i++)
        q[i] = (int)fminf(fmaxf(rintf(xn[i] * qs), -128.0f), 127.0f);

    unsigned p0 = (q[0] & 255u) | ((q[1] & 255u) << 8) | ((q[2] & 255u) << 16)
                | ((unsigned)(q[3] & 255) << 24);
    unsigned p1 = (q[4] & 255u) | ((q[5] & 255u) << 8) | ((q[6] & 255u) << 16)
                | ((unsigned)(q[7] & 255) << 24);
    v2i pk; pk[0] = (int)p0; pk[1] = (int)p1;
    *(v2i*)(xq + (size_t)row * D_IN + (size_t)t * 8) = pk;

    if (t == 0) xscale[row] = amax * (1.0f / 127.0f);
}

__global__ void __launch_bounds__(256)
bitgemm_kernel(const char* __restrict__ xq, const char* __restrict__ wq,
               const float* __restrict__ xscale, float* __restrict__ out) {
    const int lane = threadIdx.x & 31;
    const int wave = threadIdx.x >> 5;
    const int hi   = lane >> 4;
    const int l16  = lane & 15;
    const int m0   = blockIdx.y * 32;
    const int n0   = blockIdx.x * 512 + wave * 64;

    v8i c[8] = {};

    const char* pa0 = xq + (size_t)(m0 + l16) * D_IN + (hi ? 16 : 0);
    const char* pa1 = pa0 + (size_t)16 * D_IN;
    const char* pb0 = wq + (size_t)(n0 + l16) * D_IN + (hi ? 16 : 0);

    for (int k0 = 0; k0 < D_IN; k0 += 64) {
        v8i a0, a1;
        #pragma unroll
        for (int j = 0; j < 4; j++) {
            v2i t0 = *(const v2i*)(pa0 + k0 + (j >> 1) * 32 + (j & 1) * 8);
            v2i t1 = *(const v2i*)(pa1 + k0 + (j >> 1) * 32 + (j & 1) * 8);
            a0[2 * j] = t0[0]; a0[2 * j + 1] = t0[1];
            a1[2 * j] = t1[0]; a1[2 * j + 1] = t1[1];
        }
        if (k0 + 64 < D_IN) {
            __builtin_prefetch(pa0 + k0 + 64, 0, 1);
            __builtin_prefetch(pa1 + k0 + 64, 0, 1);
        }
        #pragma unroll
        for (int j = 0; j < 4; j++) {
            const char* pb = pb0 + (size_t)j * 16 * D_IN + k0;
            v4i q0 = *(const v4i*)(pb);
            v4i q1 = *(const v4i*)(pb + 32);
            v8i b;
            b[0] = q0[0]; b[1] = q0[1]; b[2] = q0[2]; b[3] = q0[3];
            b[4] = q1[0]; b[5] = q1[1]; b[6] = q1[2]; b[7] = q1[3];
            c[j]     = __builtin_amdgcn_wmma_i32_16x16x64_iu8(
                           true, a0, true, b, c[j],     false, false);
            c[4 + j] = __builtin_amdgcn_wmma_i32_16x16x64_iu8(
                           true, a1, true, b, c[4 + j], false, false);
        }
    }

    #pragma unroll
    for (int mi = 0; mi < 2; mi++) {
        #pragma unroll
        for (int r = 0; r < 8; r++) {
            const int m = m0 + mi * 16 + r + hi * 8;
            const float sc = xscale[m];
            #pragma unroll
            for (int j = 0; j < 4; j++) {
                const int n = n0 + j * 16 + l16;
                const float val = (float)(c[mi * 4 + j][r]) * sc;
                __builtin_nontemporal_store(val, out + (size_t)m * D_OUT + n);
            }
        }
    }
}

extern "C" void kernel_launch(void* const* d_in, const int* in_sizes, int n_in,
                              void* d_out, int out_size, void* d_ws, size_t ws_size,
                              hipStream_t stream) {
    const float* x     = (const float*)d_in[0];
    const float* w     = (const float*)d_in[1];
    const float* gamma = (const float*)d_in[2];
    float* out = (float*)d_out;

    char* ws = (char*)d_ws;
    float* wsum   = (float*)ws;
    float* xscale = (float*)(ws + 256);
    char*  wq     = ws + 256 + 65536;
    char*  xq     = wq + (size_t)D_OUT * D_IN;

    const int n4w = (D_OUT * D_IN) / 4;

    hipMemsetAsync(wsum, 0, sizeof(float), stream);
    wabs_reduce_kernel<<<2048, 256, 0, stream>>>((const float4*)w, wsum, n4w);
    ternarize_kernel<<<(n4w + 255) / 256, 256, 0, stream>>>(
        (const float4*)w, wsum, wq, n4w);
    rmsq_kernel<<<M_TOT, 256, 0, stream>>>(x, gamma, xq, xscale);

    dim3 grid(D_OUT / 512, M_TOT / 32);
    bitgemm_kernel<<<grid, 256, 0, stream>>>(xq, wq, xscale, out);
}
